// MultiHeadAttention_13013750907439
// MI455X (gfx1250) — matrix operand order repaired and checked by reading, not compiled, not run on hardware
//
#include <hip/hip_runtime.h>
#include <hip/hip_bf16.h>

typedef _Float16 fp16_t;
typedef _Float16 v16h __attribute__((ext_vector_type(16)));
typedef _Float16 v8h  __attribute__((ext_vector_type(8)));
typedef float    v8f  __attribute__((ext_vector_type(8)));
typedef int      v4i  __attribute__((ext_vector_type(4)));

union AFrag { v16h v; v8h h[2]; };

#define B_  2
#define S_  2048
#define C_  1024
#define NH_ 16
#define HD_ 64
#define BH_ (B_ * NH_)      // 32
#define M_  (B_ * S_)       // 4096 rows of x

// ---------------------------------------------------------------------------
// CDNA5 async global->LDS copy (ASYNCcnt-tracked), with synchronous fallback.
#if __has_builtin(__builtin_amdgcn_global_load_async_to_lds_b128)
#define HAS_ASYNC_LDS 1
#else
#define HAS_ASYNC_LDS 0
#endif

typedef __attribute__((address_space(1))) const fp16_t glb_f16;
typedef __attribute__((address_space(3))) fp16_t       lds_f16;
typedef __attribute__((address_space(1))) v4i          as1_v4i;
typedef __attribute__((address_space(3))) v4i          as3_v4i;

__device__ __forceinline__ void async_copy16(const fp16_t* g, fp16_t* l) {
#if HAS_ASYNC_LDS
    __builtin_amdgcn_global_load_async_to_lds_b128(
        (as1_v4i*)(glb_f16*)g, (as3_v4i*)(lds_f16*)l, 0, 0);
#else
    *(v8h*)l = *(const v8h*)g;
#endif
}

template <int N>
__device__ __forceinline__ void wait_async_lds() {
#if HAS_ASYNC_LDS
  #if __has_builtin(__builtin_amdgcn_s_wait_asynccnt)
    __builtin_amdgcn_s_wait_asynccnt(N);
  #else
    asm volatile("s_wait_asynccnt %0" :: "i"(N) : "memory");
  #endif
#endif
}

// ---------------------------------------------------------------------------
// f32 -> f16 convert
__global__ __launch_bounds__(256)
void cvt_f32_f16(const float* __restrict__ in, fp16_t* __restrict__ out, int n) {
    int i = blockIdx.x * blockDim.x + threadIdx.x;
    if (i < n) out[i] = (fp16_t)in[i];
}

// in[rows][cols] (f32, row-major)  ->  out[cols][rows] (f16, i.e. N-major weights)
__global__ __launch_bounds__(256)
void transpose_f32_f16(const float* __restrict__ in, fp16_t* __restrict__ out,
                       int rows, int cols) {
    int i = blockIdx.x * blockDim.x + threadIdx.x;
    if (i < rows * cols) {
        int r = i / cols, c = i % cols;
        out[(size_t)c * rows + r] = (fp16_t)in[i];
    }
}

// ---------------------------------------------------------------------------
// Tiled WMMA GEMM:  C[M,N] = A[M,K] (f16, row-major) * Bt[N,K]^T (f16, N-major) + bias
// block = 256 threads = 8 waves, 4(M) x 2(N); block tile 128 x 128; wave tile 32 x 64.
// K staged through LDS in chunks of 64, double-buffered with async global->LDS copies.
// mode 0: store f32 row-major to Cout (+bias)
// mode 1: QKV epilogue: split columns into Q/K (per-head row-major) and V (per-head
//         transposed) f16 tensors, +bias.
#define BM 128
#define BN 128
#define BK 64

__global__ __launch_bounds__(256)
void gemm_wmma(const fp16_t* __restrict__ A, const fp16_t* __restrict__ Bt,
               const float* __restrict__ bias, float* __restrict__ Cout,
               fp16_t* __restrict__ Qh, fp16_t* __restrict__ Kh,
               fp16_t* __restrict__ Vt,
               int M, int N, int K, int mode)
{
    __shared__ __align__(32) fp16_t Ash[2][BM][BK];   // 32 KB
    __shared__ __align__(32) fp16_t Bsh[2][BN][BK];   // 32 KB

    const int tid  = threadIdx.x;
    const int lane = tid & 31;
    const int w    = tid >> 5;
    const int mw   = w & 3;
    const int nw   = w >> 2;
    const int r    = lane & 15;   // row-in-tile (A/D) or col-in-tile (B/D)
    const int hi   = lane >> 4;   // half-wave selector

    const int Mblk = blockIdx.y * BM;
    const int Nblk = blockIdx.x * BN;

    // Tile copy: BM*BK halfs = 8192 = 1024 chunks of 16B per matrix; 4 chunks/thread.
    // chunk -> row = chunk>>3, 16B-slot cc = chunk&7  (BK/8 == 8 slots per row)
    auto issue_tile = [&](int buf, int k0) {
#pragma unroll
        for (int c = 0; c < 4; ++c) {
            const int chunk = tid + c * 256;
            const int row = chunk >> 3;
            const int cc  = chunk & 7;
            async_copy16(A  + (size_t)(Mblk + row) * K + k0 + cc * 8,
                         &Ash[buf][row][cc * 8]);
            async_copy16(Bt + (size_t)(Nblk + row) * K + k0 + cc * 8,
                         &Bsh[buf][row][cc * 8]);
        }
    };

    v8f acc[2][4] = {};
    const int nk = K / BK;

    issue_tile(0, 0);
    for (int kt = 0; kt < nk; ++kt) {
        const int buf = kt & 1;
        if (kt + 1 < nk) {
            issue_tile(buf ^ 1, (kt + 1) * BK);
            wait_async_lds<8>();   // 8 newest (prefetch) may stay in flight
        } else {
            wait_async_lds<0>();
        }
        __syncthreads();

#pragma unroll
        for (int ks = 0; ks < 2; ++ks) {          // two 16x16x32 steps per BK
            AFrag a[2];
#pragma unroll
            for (int mt = 0; mt < 2; ++mt) {
                const int arow = mw * 32 + mt * 16 + r;
                a[mt].h[0] = *(const v8h*)&Ash[buf][arow][ks * 32 + hi * 16];
                a[mt].h[1] = *(const v8h*)&Ash[buf][arow][ks * 32 + hi * 16 + 8];
            }
#pragma unroll
            for (int j = 0; j < 4; ++j) {
                const int brow = nw * 64 + j * 16 + r;
                v16h bfrag = *(const v16h*)&Bsh[buf][brow][ks * 32 + hi * 16];
#pragma unroll
                for (int mt = 0; mt < 2; ++mt)
                    acc[mt][j] = __builtin_amdgcn_wmma_f32_16x16x32_f16(
                        false, a[mt].v, false, bfrag, (short)0, acc[mt][j],
                        false, false);
            }
        }
        __syncthreads();
    }

#pragma unroll
    for (int mt = 0; mt < 2; ++mt) {
#pragma unroll
        for (int j = 0; j < 4; ++j) {
            const int col = Nblk + nw * 64 + j * 16 + r;
            const float bv = bias[col];
#pragma unroll
            for (int g = 0; g < 8; ++g) {
                const int row = Mblk + mw * 32 + mt * 16 + hi * 8 + g;
                const float val = acc[mt][j][g] + bv;
                if (mode == 0) {
                    Cout[(size_t)row * N + col] = val;
                } else {
                    // col = t*1024 + h*64 + d ; row = b*2048 + s
                    const int t  = col >> 10;
                    const int hd = col & 1023;
                    const int h  = hd >> 6;
                    const int d  = hd & 63;
                    const int b  = row >> 11;
                    const int s  = row & 2047;
                    const int bh = b * NH_ + h;
                    const fp16_t hv = (fp16_t)val;
                    if (t == 0)      Qh[((size_t)bh * S_ + s) * HD_ + d] = hv;
                    else if (t == 1) Kh[((size_t)bh * S_ + s) * HD_ + d] = hv;
                    else             Vt[((size_t)bh * HD_ + d) * S_ + s] = hv;
                }
            }
        }
    }
}

// ---------------------------------------------------------------------------
// Flash attention, one (b,h) per blockIdx.y, 64 query rows per block (4 waves x 16).
// Qh/Kh: [BH][S][HD] f16, Vt: [BH][HD][S] f16, Oh: [B][S][C] f16.
__global__ __launch_bounds__(128)
void attn_wmma(const fp16_t* __restrict__ Qh, const fp16_t* __restrict__ Kh,
               const fp16_t* __restrict__ Vt, fp16_t* __restrict__ Oh)
{
    __shared__ __align__(32) fp16_t psh[4][16][64];   // per-wave P staging (8 KB)

    const int lane = threadIdx.x & 31;
    const int w    = threadIdx.x >> 5;
    const int r    = lane & 15;
    const int hi   = lane >> 4;
    const int bh   = blockIdx.y;
    const int b    = bh >> 4;
    const int h    = bh & 15;
    const int qbase = blockIdx.x * 64 + w * 16;

    // Load Q A-fragments once (16 x 64, two K-steps of 32)
    AFrag qf[2];
    const fp16_t* qrow = Qh + ((size_t)bh * S_ + qbase + r) * HD_;
#pragma unroll
    for (int ks = 0; ks < 2; ++ks) {
        qf[ks].h[0] = *(const v8h*)(qrow + ks * 32 + hi * 16);
        qf[ks].h[1] = *(const v8h*)(qrow + ks * 32 + hi * 16 + 8);
    }

    float mrow[8], lrow[8];
    v8f oacc[4] = {};
#pragma unroll
    for (int g = 0; g < 8; ++g) { mrow[g] = -3.0e38f; lrow[g] = 0.0f; }

    const float scale = 0.125f; // 1/sqrt(HD)

    for (int kv = 0; kv < S_; kv += 64) {
        // ---- scores S = Q K^T : 16 x 64 in 4 tiles
        v8f sacc[4] = {};
#pragma unroll
        for (int jt = 0; jt < 4; ++jt) {
#pragma unroll
            for (int ks = 0; ks < 2; ++ks) {
                const fp16_t* kp = Kh + ((size_t)bh * S_ + kv + jt * 16 + r) * HD_
                                      + ks * 32 + hi * 16;
                v16h bfrag = *(const v16h*)kp;
                sacc[jt] = __builtin_amdgcn_wmma_f32_16x16x32_f16(
                    false, qf[ks].v, false, bfrag, (short)0, sacc[jt], false, false);
            }
        }

        // ---- online softmax (per output row = hi*8+g, spread over 16 lanes)
#pragma unroll
        for (int g = 0; g < 8; ++g) {
            float vmax = -3.0e38f;
#pragma unroll
            for (int jt = 0; jt < 4; ++jt)
                vmax = fmaxf(vmax, sacc[jt][g] * scale);
            for (int off = 1; off < 16; off <<= 1)
                vmax = fmaxf(vmax, __shfl_xor(vmax, off, 32));

            const float mnew  = fmaxf(mrow[g], vmax);
            const float alpha = __expf(mrow[g] - mnew);
            mrow[g] = mnew;

            float psum = 0.0f;
#pragma unroll
            for (int jt = 0; jt < 4; ++jt) {
                const float p = __expf(sacc[jt][g] * scale - mnew);
                psum += p;
                psh[w][hi * 8 + g][jt * 16 + r] = (fp16_t)p;
            }
            for (int off = 1; off < 16; off <<= 1)
                psum += __shfl_xor(psum, off, 32);

            lrow[g] = lrow[g] * alpha + psum;
#pragma unroll
            for (int jd = 0; jd < 4; ++jd) oacc[jd][g] *= alpha;
        }

        // wave-private LDS: ensure stores land before fragment reloads
        asm volatile("s_wait_dscnt 0" ::: "memory");

        // ---- O += P V : P is A (16 x 64 over kv), Vt rows are B fragments
#pragma unroll
        for (int ks = 0; ks < 2; ++ks) {
            AFrag pf;
            pf.h[0] = *(const v8h*)&psh[w][r][ks * 32 + hi * 16];
            pf.h[1] = *(const v8h*)&psh[w][r][ks * 32 + hi * 16 + 8];
#pragma unroll
            for (int jd = 0; jd < 4; ++jd) {
                const fp16_t* vp = Vt + ((size_t)bh * HD_ + jd * 16 + r) * S_
                                      + kv + ks * 32 + hi * 16;
                v16h vfrag = *(const v16h*)vp;
                oacc[jd] = __builtin_amdgcn_wmma_f32_16x16x32_f16(
                    false, pf.v, false, vfrag, (short)0, oacc[jd], false, false);
            }
        }
    }

    // ---- normalize and store to Oh[b][s][h*64+d] (f16, GEMM2 input)
#pragma unroll
    for (int g = 0; g < 8; ++g) {
        const float inv = 1.0f / lrow[g];
        const int s = qbase + hi * 8 + g;
#pragma unroll
        for (int jd = 0; jd < 4; ++jd) {
            const int d = jd * 16 + r;
            Oh[((size_t)b * S_ + s) * C_ + h * HD_ + d] = (fp16_t)(oacc[jd][g] * inv);
        }
    }
}

// ---------------------------------------------------------------------------
extern "C" void kernel_launch(void* const* d_in, const int* in_sizes, int n_in,
                              void* d_out, int out_size, void* d_ws, size_t ws_size,
                              hipStream_t stream)
{
    const float* x    = (const float*)d_in[0];
    const float* Wqkv = (const float*)d_in[1];
    const float* bqkv = (const float*)d_in[2];
    const float* Wo   = (const float*)d_in[3];
    const float* bo   = (const float*)d_in[4];
    float* out = (float*)d_out;

    char* ws = (char*)d_ws;
    size_t off = 0;
    auto alloc = [&](size_t bytes) -> void* {
        void* p = ws + off;
        off += (bytes + 255) & ~(size_t)255;
        return p;
    };
    fp16_t* xh    = (fp16_t*)alloc((size_t)M_ * C_ * 2);          // 8 MB
    fp16_t* wqkvt = (fp16_t*)alloc((size_t)3 * C_ * C_ * 2);      // 6 MB
    fp16_t* wot   = (fp16_t*)alloc((size_t)C_ * C_ * 2);          // 2 MB
    fp16_t* Qh    = (fp16_t*)alloc((size_t)BH_ * S_ * HD_ * 2);   // 8 MB
    fp16_t* Kh    = (fp16_t*)alloc((size_t)BH_ * S_ * HD_ * 2);   // 8 MB
    fp16_t* Vt    = (fp16_t*)alloc((size_t)BH_ * S_ * HD_ * 2);   // 8 MB
    fp16_t* Oh    = (fp16_t*)alloc((size_t)M_ * C_ * 2);          // 8 MB

    // 1) precision conversion / weight transposition
    cvt_f32_f16<<<(M_ * C_) / 256, 256, 0, stream>>>(x, xh, M_ * C_);
    transpose_f32_f16<<<(C_ * 3 * C_) / 256, 256, 0, stream>>>(Wqkv, wqkvt, C_, 3 * C_);
    transpose_f32_f16<<<(C_ * C_) / 256, 256, 0, stream>>>(Wo, wot, C_, C_);

    // 2) QKV projection (fused bias + head split + V transpose)
    gemm_wmma<<<dim3((3 * C_) / BN, M_ / BM), 256, 0, stream>>>(
        xh, wqkvt, bqkv, nullptr, Qh, Kh, Vt, M_, 3 * C_, C_, 1);

    // 3) flash attention
    attn_wmma<<<dim3(S_ / 64, BH_), 128, 0, stream>>>(Qh, Kh, Vt, Oh);

    // 4) output projection -> f32 d_out
    gemm_wmma<<<dim3(C_ / BN, M_ / BM), 256, 0, stream>>>(
        Oh, wot, bo, out, nullptr, nullptr, nullptr, M_, C_, C_, 0);

    (void)in_sizes; (void)n_in; (void)out_size; (void)ws_size;
}
